// Relation_66683662238343
// MI455X (gfx1250) — matrix operand order repaired and checked by reading, not compiled, not run on hardware
//
#include <hip/hip_runtime.h>
#include <hip/hip_bf16.h>

#define Sdim 512
#define Hdim 512
#define Rdim 64
#define Bdim 4
#define RELSTRIDE 516   // 516 f32: conflict-free column writes (s*516 mod 64 = 4s)

typedef __attribute__((ext_vector_type(16))) __bf16 bf16x16;
typedef __attribute__((ext_vector_type(8)))  float  f32x8;

union Frag {
    bf16x16 v;
    unsigned int u[8];
};

__device__ __forceinline__ unsigned int pk_bf16(float a, float b) {
    unsigned int ua = __float_as_uint(a), ub = __float_as_uint(b);
    ua = (ua + 0x7FFFu + ((ua >> 16) & 1u)) >> 16;
    ub = (ub + 0x7FFFu + ((ub >> 16) & 1u)) >> 16;
    return (ua & 0xFFFFu) | (ub << 16);
}
__device__ __forceinline__ unsigned short bf16_1(float a) {
    unsigned int ua = __float_as_uint(a);
    ua = (ua + 0x7FFFu + ((ua >> 16) & 1u)) >> 16;
    return (unsigned short)ua;
}
__device__ __forceinline__ f32x8 wmma_bf16(const Frag& a, const Frag& b, f32x8 c) {
    return __builtin_amdgcn_wmma_f32_16x16x32_bf16(false, a.v, false, b.v, (short)0, c, false, false);
}

// ---------------- LayerNorm -> x_bf16 -----------------------------------
__global__ void __launch_bounds__(256) ln_kernel(const float* __restrict__ node,
                                                 const float* __restrict__ g,
                                                 const float* __restrict__ bb,
                                                 unsigned int* __restrict__ xbf) {
    __shared__ float sred[256];
    const int row = blockIdx.x, t = threadIdx.x;
    const float2 xv = *(const float2*)(node + (size_t)row * Hdim + 2 * t);
    float a = xv.x, b2 = xv.y;
    sred[t] = a + b2; __syncthreads();
    for (int s = 128; s > 0; s >>= 1) { if (t < s) sred[t] += sred[t + s]; __syncthreads(); }
    const float mean = sred[0] * (1.0f / Hdim);
    __syncthreads();
    const float da = a - mean, db = b2 - mean;
    sred[t] = da * da + db * db; __syncthreads();
    for (int s = 128; s > 0; s >>= 1) { if (t < s) sred[t] += sred[t + s]; __syncthreads(); }
    const float rstd = rsqrtf(sred[0] * (1.0f / Hdim) + 1e-5f);
    const float ya = da * rstd * g[2 * t] + bb[2 * t];
    const float yb = db * rstd * g[2 * t + 1] + bb[2 * t + 1];
    xbf[(size_t)row * (Hdim / 2) + t] = pk_bf16(ya, yb);
}

// ---------------- Generic bf16 WMMA GEMM: out[m,n] = A[m,:]·W[n,:] + bias[n]
// A: bf16 [M x K] row-major, W: f32 [N x K] row-major. 64x64 tile / WG (8 waves).
__global__ void __launch_bounds__(256) gemm_bf16_kernel(const unsigned short* __restrict__ A,
                                                        const float* __restrict__ W,
                                                        const float* __restrict__ bias,
                                                        float* __restrict__ out,
                                                        int N, int K) {
    __shared__ unsigned short lA[64 * 40];   // 32 data + 8 pad per row
    __shared__ unsigned short lB[64 * 40];
    const int t = threadIdx.x;
    const int wave = t >> 5, lane = t & 31, half = lane >> 4, l16 = lane & 15;
    const int mt = wave & 3, ntb = (wave >> 2) * 2;
    const int m0 = blockIdx.y * 64, n0 = blockIdx.x * 64;
    const int crow = t >> 2, cchunk = t & 3;
    f32x8 acc0 = {}; f32x8 acc1 = {};
    for (int k0 = 0; k0 < K; k0 += 32) {
        __syncthreads();
        {   // A tile (already bf16): 16B per thread
            const uint4 av = *(const uint4*)(A + (size_t)(m0 + crow) * K + k0 + cchunk * 8);
            *(uint4*)(lA + crow * 40 + cchunk * 8) = av;
        }
        {   // W tile: convert f32 -> bf16
            const float4* ws = (const float4*)(W + (size_t)(n0 + crow) * K + k0 + cchunk * 8);
            const float4 w0 = ws[0], w1 = ws[1];
            uint4 p;
            p.x = pk_bf16(w0.x, w0.y); p.y = pk_bf16(w0.z, w0.w);
            p.z = pk_bf16(w1.x, w1.y); p.w = pk_bf16(w1.z, w1.w);
            *(uint4*)(lB + crow * 40 + cchunk * 8) = p;
        }
        __syncthreads();
        Frag aF, bF0, bF1;
        const unsigned short* arow = lA + (mt * 16 + l16) * 40;
#pragma unroll
        for (int e = 0; e < 8; ++e) {
            const int kk = half * 16 + e * 2;
            aF.u[e] = *(const unsigned int*)(arow + kk);
        }
        {
            const uint4* bp = (const uint4*)(lB + (ntb * 16 + l16) * 40 + half * 16);
            const uint4 q0 = bp[0], q1 = bp[1];
            bF0.u[0] = q0.x; bF0.u[1] = q0.y; bF0.u[2] = q0.z; bF0.u[3] = q0.w;
            bF0.u[4] = q1.x; bF0.u[5] = q1.y; bF0.u[6] = q1.z; bF0.u[7] = q1.w;
        }
        {
            const uint4* bp = (const uint4*)(lB + ((ntb + 1) * 16 + l16) * 40 + half * 16);
            const uint4 q0 = bp[0], q1 = bp[1];
            bF1.u[0] = q0.x; bF1.u[1] = q0.y; bF1.u[2] = q0.z; bF1.u[3] = q0.w;
            bF1.u[4] = q1.x; bF1.u[5] = q1.y; bF1.u[6] = q1.z; bF1.u[7] = q1.w;
        }
        acc0 = wmma_bf16(aF, bF0, acc0);
        acc1 = wmma_bf16(aF, bF1, acc1);
    }
    const int n_a = n0 + ntb * 16 + l16, n_b = n_a + 16;
    const float bia = bias[n_a], bib = bias[n_b];
    const int mbase = m0 + mt * 16 + half * 8;
#pragma unroll
    for (int vv = 0; vv < 8; ++vv) {
        out[(size_t)(mbase + vv) * N + n_a] = acc0[vv] + bia;
        out[(size_t)(mbase + vv) * N + n_b] = acc1[vv] + bib;
    }
}

// ---------------- Main fused relation kernel: one WG per (b,i) -----------
// rel2^T[s,j] = Wt @ (q*k + We@edge^T + Wp@path^T) + mask; softmax over j;
// out[r,d] = sum_j attn[j,r] * v[j,r,d]  -> attnout bf16.
#define MAIN_LDS_BYTES 181760

__global__ void __launch_bounds__(256) relation_main_kernel(
        const float* __restrict__ edge, const float* __restrict__ path,
        const float* __restrict__ mask,
        const float* __restrict__ gWe, const float* __restrict__ gWp, const float* __restrict__ gWt,
        const float* __restrict__ q_ws, const float* __restrict__ k_ws,
        const float* __restrict__ v_ws, unsigned int* __restrict__ attnout) {
    extern __shared__ char smem[];
    unsigned short* lwe = (unsigned short*)smem;          // 4096
    unsigned short* lwp = lwe + 4096;                     // 4096
    unsigned short* lwt = lwp + 4096;                     // 4096
    unsigned short* stageAll = lwt + 4096;                // 8 waves * 16*72
    float* qsh  = (float*)(stageAll + 8 * 1152);          // 64
    float* redm = qsh + 64;                               // 256
    float* reds = redm + 256;                             // 256
    float* rinv = reds + 256;                             // 64
    float* pbuf = rinv + 64;                              // 1024
    float* rel  = pbuf + 1024;                            // 64*516

    const int t = threadIdx.x;
    const int wave = t >> 5, lane = t & 31, half = lane >> 4, l16 = lane & 15;
    const int bi = blockIdx.x;
    const int b = bi >> 9;

    // ---- phase 0: weights -> LDS bf16, q -> LDS
    for (int idx = t; idx < 4096; idx += 256) {
        lwe[idx] = bf16_1(gWe[idx]);
        lwp[idx] = bf16_1(gWp[idx]);
        lwt[idx] = bf16_1(gWt[idx]);
    }
    if (t < 64) qsh[t] = q_ws[(size_t)bi * 64 + t];
    __syncthreads();

    // ---- persistent A-fragments of We/Wp/Wt (reused for all 32 j-tiles)
    Frag aWe[4][2], aWp[4][2], aWt[4][2];
#pragma unroll
    for (int mt = 0; mt < 4; ++mt)
#pragma unroll
        for (int kk = 0; kk < 2; ++kk)
#pragma unroll
            for (int e = 0; e < 8; ++e) {
                const int off = (mt * 16 + l16) * 64 + kk * 32 + half * 16 + e * 2;
                aWe[mt][kk].u[e] = *(const unsigned int*)(lwe + off);
                aWp[mt][kk].u[e] = *(const unsigned int*)(lwp + off);
                aWt[mt][kk].u[e] = *(const unsigned int*)(lwt + off);
            }
    float q8[4][8];
#pragma unroll
    for (int mt = 0; mt < 4; ++mt)
#pragma unroll
        for (int vv = 0; vv < 8; ++vv)
            q8[mt][vv] = qsh[mt * 16 + half * 8 + vv];

    unsigned short* stg = stageAll + wave * 1152;  // [j(16)][r(64)+8 pad] bf16

    // ---- phase 1: per j-tile GEMMs + rank-1 + Wt + mask -> rel[s][j]
    for (int jt = wave; jt < 32; jt += 8) {
        const int j0 = jt * 16;
        const int j = j0 + l16;
        const float* erow = edge + ((size_t)bi * Sdim + j) * Rdim + half * 16;
        const float* prow = path + ((size_t)bi * Sdim + j) * Rdim + half * 16;
        if (jt + 8 < 32) {
            __builtin_prefetch(erow + (size_t)8 * 16 * Rdim, 0, 0);
            __builtin_prefetch(prow + (size_t)8 * 16 * Rdim, 0, 0);
        }
        Frag bE[2], bP[2];
#pragma unroll
        for (int kk = 0; kk < 2; ++kk) {
            const float4* e4 = (const float4*)(erow + kk * 32);
            const float4* p4 = (const float4*)(prow + kk * 32);
#pragma unroll
            for (int c = 0; c < 4; ++c) {
                const float4 ev = e4[c], pv = p4[c];
                bE[kk].u[2 * c]     = pk_bf16(ev.x, ev.y);
                bE[kk].u[2 * c + 1] = pk_bf16(ev.z, ev.w);
                bP[kk].u[2 * c]     = pk_bf16(pv.x, pv.y);
                bP[kk].u[2 * c + 1] = pk_bf16(pv.z, pv.w);
            }
        }
        const float mval = mask[(size_t)bi * Sdim + j];
        const float* krow = k_ws + (size_t)(b * Sdim + j) * Rdim;

#pragma unroll
        for (int mt = 0; mt < 4; ++mt) {
            f32x8 C = {};
            C = wmma_bf16(aWe[mt][0], bE[0], C);
            C = wmma_bf16(aWe[mt][1], bE[1], C);
            C = wmma_bf16(aWp[mt][0], bP[0], C);
            C = wmma_bf16(aWp[mt][1], bP[1], C);
            const int rbase = mt * 16 + half * 8;
            const float4* k4 = (const float4*)(krow + rbase);
            const float4 k0 = k4[0], k1 = k4[1];
            C[0] += q8[mt][0] * k0.x; C[1] += q8[mt][1] * k0.y;
            C[2] += q8[mt][2] * k0.z; C[3] += q8[mt][3] * k0.w;
            C[4] += q8[mt][4] * k1.x; C[5] += q8[mt][5] * k1.y;
            C[6] += q8[mt][6] * k1.z; C[7] += q8[mt][7] * k1.w;
            unsigned int* sp = (unsigned int*)(stg + l16 * 72 + rbase);
            sp[0] = pk_bf16(C[0], C[1]); sp[1] = pk_bf16(C[2], C[3]);
            sp[2] = pk_bf16(C[4], C[5]); sp[3] = pk_bf16(C[6], C[7]);
        }
        asm volatile("" ::: "memory");   // LDS is in-order within a wave

        Frag bT[2];
#pragma unroll
        for (int kk = 0; kk < 2; ++kk) {
            const uint4* tp = (const uint4*)(stg + l16 * 72 + kk * 32 + half * 16);
            const uint4 q0 = tp[0], q1 = tp[1];
            bT[kk].u[0] = q0.x; bT[kk].u[1] = q0.y; bT[kk].u[2] = q0.z; bT[kk].u[3] = q0.w;
            bT[kk].u[4] = q1.x; bT[kk].u[5] = q1.y; bT[kk].u[6] = q1.z; bT[kk].u[7] = q1.w;
        }
#pragma unroll
        for (int mt = 0; mt < 4; ++mt) {
            f32x8 C = {};
            C = wmma_bf16(aWt[mt][0], bT[0], C);
            C = wmma_bf16(aWt[mt][1], bT[1], C);
            const int sbase = mt * 16 + half * 8;
#pragma unroll
            for (int vv = 0; vv < 8; ++vv)
                rel[(sbase + vv) * RELSTRIDE + j0 + l16] = C[vv] + mval;
        }
    }
    __syncthreads();

    // ---- phase 2: softmax over j per s (4 threads per row)
    {
        const int s = t & 63, quarter = t >> 6;
        float* rr = rel + s * RELSTRIDE + quarter * 128;
        float m = -3.4e38f;
        for (int jj = 0; jj < 128; ++jj) m = fmaxf(m, rr[jj]);
        redm[quarter * 64 + s] = m;
        __syncthreads();
        m = fmaxf(fmaxf(redm[s], redm[64 + s]), fmaxf(redm[128 + s], redm[192 + s]));
        float lsum = 0.f;
        for (int jj = 0; jj < 128; ++jj) { const float e = __expf(rr[jj] - m); rr[jj] = e; lsum += e; }
        reds[quarter * 64 + s] = lsum;
        __syncthreads();
        if (t < 64) rinv[t] = 1.0f / (reds[t] + reds[64 + t] + reds[128 + t] + reds[192 + t]);
        __syncthreads();
    }

    // ---- phase 3: out[r,d] = (1/sum) * sum_j e[r][j] * v[b,j,r,d]
    {
        const int r = (t >> 1) & 63, db = (t & 1) * 4, jh = t >> 7;
        const float* er = rel + r * RELSTRIDE + jh * 256;
        const float4* vp = (const float4*)(v_ws + ((size_t)(b * Sdim) + jh * 256) * Hdim + r * 8 + db);
        float4 acc = {0.f, 0.f, 0.f, 0.f};
        for (int jj = 0; jj < 256; ++jj) {
            const float e = er[jj];
            const float4 vv = *vp;
            acc.x += e * vv.x; acc.y += e * vv.y; acc.z += e * vv.z; acc.w += e * vv.w;
            vp += Hdim / 4;
        }
        *(float4*)(pbuf + jh * 512 + r * 8 + db) = acc;
        __syncthreads();
        if (t < 128) {
            const int idx = t * 4;
            const float4 p0 = *(float4*)(pbuf + idx);
            const float4 p1 = *(float4*)(pbuf + 512 + idx);
            const float sc = rinv[t >> 1];
            attnout[(size_t)bi * 256 + t * 2]     = pk_bf16((p0.x + p1.x) * sc, (p0.y + p1.y) * sc);
            attnout[(size_t)bi * 256 + t * 2 + 1] = pk_bf16((p0.z + p1.z) * sc, (p0.w + p1.w) * sc);
        }
    }
}

// ------------------------------------------------------------------------
extern "C" void kernel_launch(void* const* d_in, const int* in_sizes, int n_in,
                              void* d_out, int out_size, void* d_ws, size_t ws_size,
                              hipStream_t stream) {
    (void)in_sizes; (void)n_in; (void)out_size; (void)ws_size;
    const float* node = (const float*)d_in[0];
    const float* edge = (const float*)d_in[1];
    const float* path = (const float*)d_in[2];
    const float* rmsk = (const float*)d_in[3];
    const float* ln_g = (const float*)d_in[4];
    const float* ln_b = (const float*)d_in[5];
    const float* Wq   = (const float*)d_in[6];
    const float* bq   = (const float*)d_in[7];
    const float* Wk   = (const float*)d_in[8];
    const float* bk   = (const float*)d_in[9];
    const float* Wv   = (const float*)d_in[10];
    const float* bv   = (const float*)d_in[11];
    const float* We   = (const float*)d_in[12];
    const float* Wp   = (const float*)d_in[13];
    const float* Wt   = (const float*)d_in[14];
    const float* Wo   = (const float*)d_in[15];
    const float* bo   = (const float*)d_in[16];
    float* out = (float*)d_out;

    const int M = Bdim * Sdim;                    // 2048 rows
    char* ws = (char*)d_ws;
    unsigned short* xbf = (unsigned short*)ws;                        // M*H bf16
    float* q_ws = (float*)(ws + (size_t)M * Hdim * 2);
    float* k_ws = q_ws + (size_t)M * Rdim;
    float* v_ws = k_ws + (size_t)M * Rdim;
    unsigned short* attnout = (unsigned short*)(v_ws + (size_t)M * Hdim);

    ln_kernel<<<M, 256, 0, stream>>>(node, ln_g, ln_b, (unsigned int*)xbf);
    gemm_bf16_kernel<<<dim3(1, M / 64), 256, 0, stream>>>(xbf, Wq, bq, q_ws, Rdim, Hdim);
    gemm_bf16_kernel<<<dim3(1, M / 64), 256, 0, stream>>>(xbf, Wk, bk, k_ws, Rdim, Hdim);
    gemm_bf16_kernel<<<dim3(Hdim / 64, M / 64), 256, 0, stream>>>(xbf, Wv, bv, v_ws, Hdim, Hdim);

    hipFuncSetAttribute((const void*)relation_main_kernel,
                        hipFuncAttributeMaxDynamicSharedMemorySize, MAIN_LDS_BYTES);
    relation_main_kernel<<<M, 256, MAIN_LDS_BYTES, stream>>>(
        edge, path, rmsk, We, Wp, Wt, q_ws, k_ws, v_ws, (unsigned int*)attnout);

    gemm_bf16_kernel<<<dim3(Hdim / 64, M / 64), 256, 0, stream>>>(attnout, Wo, bo, out, Hdim, Hdim);
}
